// DiscretePolicy_21217138442482
// MI455X (gfx1250) — matrix operand order repaired and checked by reading, not compiled, not run on hardware
//
#include <hip/hip_runtime.h>
#include <hip/hip_bf16.h>

// ---------------------------------------------------------------------------
// Problem constants (from the reference)
// ---------------------------------------------------------------------------
#define INPUT_DIM   128
#define HIDDEN      128
#define OUT_DIM     64
#define HYPER_H     1024
#define TOTAL_PARAMS 57792   // 128*128+128 + 2*(128*128+128) + 128*64+64

typedef __bf16 v16bf __attribute__((ext_vector_type(16)));
typedef float  v8f   __attribute__((ext_vector_type(8)));

union FragU { int4 i4[2]; v16bf v; };

// LDS flat pointer -> 32-bit DS offset (standard amdgcn truncation)
__device__ __forceinline__ unsigned lds_addr(const void* p) {
    return (unsigned)(unsigned long long)p;
}

// ---------------------------------------------------------------------------
// fp32 -> bf16 conversion, vectorized: float4 in -> 4x bf16 (8B) out
// ---------------------------------------------------------------------------
struct __align__(8) BF4 { __hip_bfloat16 a, b, c, d; };

__global__ void f32_to_bf16_kernel(const float4* __restrict__ src,
                                   BF4* __restrict__ dst,
                                   long long n4) {
    long long i = (long long)blockIdx.x * blockDim.x + threadIdx.x;
    long long stride = (long long)gridDim.x * blockDim.x;
    for (; i < n4; i += stride) {
        float4 f = src[i];
        BF4 o;
        o.a = __float2bfloat16(f.x);
        o.b = __float2bfloat16(f.y);
        o.c = __float2bfloat16(f.z);
        o.d = __float2bfloat16(f.w);
        dst[i] = o;
    }
}

// ---------------------------------------------------------------------------
// h1 = relu(ow @ W1 + b1) : (B,3) x (3,1024) -> (B,1024) bf16.  K=3 -> VALU.
// ---------------------------------------------------------------------------
__global__ void hyper1_kernel(const float* __restrict__ ow,
                              const float* __restrict__ W1,
                              const float* __restrict__ b1,
                              __hip_bfloat16* __restrict__ h1,
                              int B) {
    int idx = blockIdx.x * blockDim.x + threadIdx.x;
    if (idx >= B * HYPER_H) return;
    int b = idx >> 10;
    int j = idx & (HYPER_H - 1);
    float v = b1[j]
            + ow[b * 3 + 0] * W1[0 * HYPER_H + j]
            + ow[b * 3 + 1] * W1[1 * HYPER_H + j]
            + ow[b * 3 + 2] * W1[2 * HYPER_H + j];
    h1[idx] = __float2bfloat16(fmaxf(v, 0.f));
}

// ---------------------------------------------------------------------------
// bf16 WMMA GEMM:  C(bf16) = act( A(MxK,bf16) @ B(KxN,bf16) + bias(f32) )
// BM=256, BN=64, BK=32.  256 threads = 8 waves; wave w owns rows [32w,32w+32)
// x all 64 cols => 2 A-frags x 4 B-frags = 8 v_wmma_f32_16x16x32_bf16 / step.
// A tile staged with global_load_async_to_lds_b128 (ASYNCcnt, zero VGPR
// payload); B tile packed to "K-pair" dwords via a 4-dword register path.
// Double-buffered LDS.  Requires: M%256==0, N%64==0, K%32==0.
// ---------------------------------------------------------------------------
__global__ __launch_bounds__(256)
void gemm_bf16_relu_kernel(const __hip_bfloat16* __restrict__ A,
                           const __hip_bfloat16* __restrict__ Bm,
                           const float* __restrict__ bias,
                           __hip_bfloat16* __restrict__ C,
                           int M, int N, int K, int doRelu) {
    // A tile: 256 rows x 32 bf16, row stride 20 dwords (padded, 16B aligned)
    // B tile: "K-pair" layout: dword[n*20 + p] = (B[2p][n], B[2p+1][n])
    __shared__ __align__(16) unsigned int ldsA[2][256 * 20];  // 40 KB
    __shared__ __align__(16) unsigned int ldsB[2][64 * 20];   // 10 KB

    const int tid  = threadIdx.x;
    const int lane = tid & 31;
    const int wave = tid >> 5;
    const int half = lane >> 4;
    const int l16  = lane & 15;

    const int rowBlk = blockIdx.y * 256;
    const int colBlk = blockIdx.x * 64;

    // ---- A staging: 1024 async B128s (256 rows x 4), 4 per thread ----------
    const int aRow = tid >> 2;
    const int aC4  = tid & 3;
    const __hip_bfloat16* aPtr = A + (size_t)(rowBlk + aRow) * K + aC4 * 8;
    const size_t aSstride = (size_t)64 * K;                   // +64 rows per s
    const unsigned aLdsByte = lds_addr(&ldsA[0][0]) + 4u * (unsigned)(aRow * 20 + aC4 * 4);
    const unsigned aBufStep = (unsigned)sizeof(ldsA[0]);      // 20480 B

    auto stageA = [&](int bb) {                 // global -> LDS, no VGPR payload
        const unsigned base = aLdsByte + (unsigned)bb * aBufStep;
#pragma unroll
        for (int s = 0; s < 4; ++s) {
            const void* g = (const void*)(aPtr + aSstride * (size_t)s);
            unsigned l = base + (unsigned)(s * 64 * 20 * 4);
            asm volatile("global_load_async_to_lds_b128 %0, %1, off"
                         :: "v"(l), "v"(g) : "memory");
        }
        aPtr += 32;
    };

    // ---- B staging: pack K-pairs through 4 dword registers ------------------
    const int bP   = tid >> 5;                                // 0..7 (s adds 8)
    const int bNcp = tid & 31;
    const __hip_bfloat16* bPtr = Bm + (size_t)(2 * bP) * N + colBlk + bNcp * 2;
    const size_t bSstride = (size_t)16 * N;
    const size_t bKstride = (size_t)32 * N;
    const int bLds0 = (2 * bNcp + 0) * 20 + bP;
    const int bLds1 = (2 * bNcp + 1) * 20 + bP;

    unsigned rb0[2], rb1[2];
    auto loadB = [&]() {
#pragma unroll
        for (int s = 0; s < 2; ++s) {
            const __hip_bfloat16* bp = bPtr + bSstride * (size_t)s;
            rb0[s] = *(const unsigned*)bp;
            rb1[s] = *(const unsigned*)(bp + N);
        }
        bPtr += bKstride;
    };
    auto storeB = [&](int bb) {
#pragma unroll
        for (int s = 0; s < 2; ++s) {
            unsigned w0 = rb0[s], w1 = rb1[s];
            ldsB[bb][bLds0 + s * 8] = (w0 & 0xffffu) | (w1 << 16);
            ldsB[bb][bLds1 + s * 8] = (w0 >> 16) | (w1 & 0xffff0000u);
        }
    };

    v8f acc0[4], acc1[4];
#pragma unroll
    for (int t = 0; t < 4; ++t)
#pragma unroll
        for (int r = 0; r < 8; ++r) { acc0[t][r] = 0.f; acc1[t][r] = 0.f; }

    // ---- pipeline ----------------------------------------------------------
    stageA(0);
    loadB();
    storeB(0);
    asm volatile("s_wait_asynccnt 0x0" ::: "memory");
    __syncthreads();

    const int ksteps = K >> 5;
    for (int kt = 0; kt < ksteps; ++kt) {
        const int cur = kt & 1;
        const bool more = (kt + 1 < ksteps);
        if (more) {                          // next tile in flight during WMMAs
            stageA(1 - cur);
            loadB();
        }

        const unsigned int* lA = ldsA[cur];
        const unsigned int* lB = ldsB[cur];
        FragU fa0, fa1;
        {
            const int ar = (wave * 32 + l16) * 20;
            fa0.i4[0] = *(const int4*)&lA[ar + 4 * half];
            fa0.i4[1] = *(const int4*)&lA[ar + 8 + 4 * half];
            fa1.i4[0] = *(const int4*)&lA[ar + 320 + 4 * half];       // +16 rows
            fa1.i4[1] = *(const int4*)&lA[ar + 320 + 8 + 4 * half];
        }
#pragma unroll
        for (int t = 0; t < 4; ++t) {
            FragU fbt;
            const int nb = (t * 16 + l16) * 20 + 4 * half;
            fbt.i4[0] = *(const int4*)&lB[nb];
            fbt.i4[1] = *(const int4*)&lB[nb + 8];
            acc0[t] = __builtin_amdgcn_wmma_f32_16x16x32_bf16(
                false, fa0.v, false, fbt.v, (short)0, acc0[t], false, false);
            acc1[t] = __builtin_amdgcn_wmma_f32_16x16x32_bf16(
                false, fa1.v, false, fbt.v, (short)0, acc1[t], false, false);
        }

        if (more) storeB(1 - cur);
        asm volatile("s_wait_asynccnt 0x0" ::: "memory");
        __syncthreads();
    }

    // ---- epilogue: bias + optional relu, write bf16 ------------------------
#pragma unroll
    for (int t = 0; t < 4; ++t) {
        const int n = colBlk + t * 16 + l16;
        const float bv = bias[n];
#pragma unroll
        for (int r = 0; r < 8; ++r) {
            const int m0 = rowBlk + wave * 32 + half * 8 + r;
            float v0 = acc0[t][r] + bv;
            float v1 = acc1[t][r] + bv;
            if (doRelu) { v0 = fmaxf(v0, 0.f); v1 = fmaxf(v1, 0.f); }
            C[(size_t)m0 * N + n]        = __float2bfloat16(v0);
            C[(size_t)(m0 + 16) * N + n] = __float2bfloat16(v1);
        }
    }
}

// ---------------------------------------------------------------------------
// Per-sample target MLP + softmax.  One 128-thread block per sample.
// Weights staged coalesced into LDS (row stride 69 dwords -> conflict-free),
// consumed as packed dwords; bf16->f32 via 16-bit shift (__uint_as_float).
// ---------------------------------------------------------------------------
__global__ __launch_bounds__(128)
void target_forward_kernel(const float* __restrict__ x,
                           const __hip_bfloat16* __restrict__ flat,
                           float* __restrict__ out) {
    __shared__ __align__(4) unsigned int Wlu[128 * 69];   // 35.3 KB
    __shared__ float actA[128];
    __shared__ float actB[128];

    const int b   = blockIdx.x;
    const int tid = threadIdx.x;
    const __hip_bfloat16* fb = flat + (size_t)b * TOTAL_PARAMS;
    const unsigned int* fbu  = (const unsigned int*)fb;   // b*57792*2B is 4B aligned

    actA[tid] = x[(size_t)b * INPUT_DIM + tid];

    const int wOff[4] = {0, 16512, 33024, 49536};
    const int bOff[4] = {16384, 32896, 49408, 57728};
    const int nOut[4] = {HIDDEN, HIDDEN, HIDDEN, OUT_DIM};

    float* cur = actA;
    float* nxt = actB;
#pragma unroll 1
    for (int l = 0; l < 4; ++l) {
        const int no = nOut[l];
        const int dw = no * 64;              // #dwords of weights this layer
        __syncthreads();                     // protect Wlu/acts from prior layer
        for (int idx = tid; idx < dw; idx += 128) {
            unsigned int w = fbu[(wOff[l] >> 1) + idx];  // coalesced
            Wlu[(idx >> 6) * 69 + (idx & 63)] = w;       // row o, dword i/2
        }
        float bv = 0.f;
        if (tid < no) bv = __bfloat162float(fb[bOff[l] + tid]);
        __syncthreads();
        if (tid < no) {
            float acc = bv;
            const unsigned int* wr = Wlu + tid * 69;
#pragma unroll 16
            for (int i2 = 0; i2 < 64; ++i2) {
                unsigned int w = wr[i2];
                acc = fmaf(__uint_as_float(w << 16),         cur[2 * i2 + 0], acc);
                acc = fmaf(__uint_as_float(w & 0xffff0000u), cur[2 * i2 + 1], acc);
            }
            if (l < 3) acc = fmaxf(acc, 0.f);
            nxt[tid] = acc;
        }
        float* tswap = cur; cur = nxt; nxt = tswap;
    }

    __syncthreads();
    if (tid < OUT_DIM) {
        float mx = -3.402823466e38f;
        for (int i = 0; i < OUT_DIM; ++i) mx = fmaxf(mx, cur[i]);
        float s = 0.f;
        for (int i = 0; i < OUT_DIM; ++i) s += __expf(cur[i] - mx);
        out[(size_t)b * OUT_DIM + tid] = __expf(cur[tid] - mx) / s;
    }
}

// ---------------------------------------------------------------------------
// Launch
// ---------------------------------------------------------------------------
extern "C" void kernel_launch(void* const* d_in, const int* in_sizes, int n_in,
                              void* d_out, int out_size, void* d_ws, size_t ws_size,
                              hipStream_t stream) {
    const float* x   = (const float*)d_in[0];  // (B,128)
    const float* ow  = (const float*)d_in[1];  // (B,3)
    const float* W1  = (const float*)d_in[2];  // (3,1024)
    const float* b1  = (const float*)d_in[3];  // (1024)
    const float* W2  = (const float*)d_in[4];  // (1024,1024)
    const float* b2  = (const float*)d_in[5];  // (1024)
    const float* W3  = (const float*)d_in[6];  // (1024,57792)
    const float* b3  = (const float*)d_in[7];  // (57792)
    float* out = (float*)d_out;

    const int B = in_sizes[0] / INPUT_DIM;     // 2048

    // Workspace layout (bf16): h1 | h2 | W2b | W3b | flat  (~349 MB total)
    __hip_bfloat16* h1   = (__hip_bfloat16*)d_ws;
    __hip_bfloat16* h2   = h1  + (size_t)B * HYPER_H;
    __hip_bfloat16* W2b  = h2  + (size_t)B * HYPER_H;
    __hip_bfloat16* W3b  = W2b + (size_t)HYPER_H * HYPER_H;
    __hip_bfloat16* flat = W3b + (size_t)HYPER_H * TOTAL_PARAMS;

    // 1) convert hyper weights to bf16 (W3b fits in 192MB L2 afterwards)
    f32_to_bf16_kernel<<<1024, 256, 0, stream>>>(
        (const float4*)W2, (BF4*)W2b, (long long)(HYPER_H * HYPER_H) / 4);
    f32_to_bf16_kernel<<<8192, 256, 0, stream>>>(
        (const float4*)W3, (BF4*)W3b, ((long long)HYPER_H * TOTAL_PARAMS) / 4);

    // 2) h1 = relu(ow @ W1 + b1)
    hyper1_kernel<<<(B * HYPER_H + 255) / 256, 256, 0, stream>>>(ow, W1, b1, h1, B);

    // 3) h2 = relu(h1 @ W2 + b2)   [2048 x 1024 x 1024]
    dim3 g2(HYPER_H / 64, B / 256);
    gemm_bf16_relu_kernel<<<g2, 256, 0, stream>>>(h1, W2b, b2, h2, B, HYPER_H, HYPER_H, 1);

    // 4) flat = h2 @ W3 + b3       [2048 x 1024 x 57792]  -> 242 GFLOP WMMA
    dim3 g3(TOTAL_PARAMS / 64, B / 256);
    gemm_bf16_relu_kernel<<<g3, 256, 0, stream>>>(h2, W3b, b3, flat, B, TOTAL_PARAMS, HYPER_H, 0);

    // 5) per-sample target MLP + softmax
    target_forward_kernel<<<B, 128, 0, stream>>>(x, flat, out);
}
